// RecurrentRGCN_1142461300899
// MI455X (gfx1250) — compile-verified, matrix operand order checked by reading, not run on hardware
//
#include <hip/hip_runtime.h>

// ---------------------------------------------------------------------------
// RecurrentRGCN for MI455X (gfx1250, wave32, WMMA + Tensor Data Mover).
// Dense GEMMs: v_wmma_f32_16x16x32_f16, f16 operands zero-padded in scratch.
// Weight panel (64 N-rows x Kp) staged into LDS once per block via
// tensor_load_to_lds (TENSORcnt), B fragments then come from ds_load_b128.
// Graph aggregation = f32 atomic scatter-add. Requires ws_size >= ~1.4 GB.
// ---------------------------------------------------------------------------

#define N_ENT   100000
#define R2C     500
#define HH      200
#define T_STEPS 3
#define E_EDGES 300000
#define R2E_C   400000
#define SE_C    4000
#define SR_C    8
#define S2E_C   8000

// Padded dims: every N/K dim is a multiple of 64 so blocks own exact panels.
#define HP   256
#define H2P  512
#define H4   832
#define H3P  640
#define R2P  512
#define SRP  16

#define RRELU_SLOPE 0.22916666666666666f   // (1/8 + 1/3) / 2

typedef __attribute__((ext_vector_type(16))) _Float16 v16h;
typedef __attribute__((ext_vector_type(8)))  _Float16 v8h;
typedef __attribute__((ext_vector_type(8)))  float    v8f;
typedef __attribute__((ext_vector_type(4)))  unsigned v4u;
typedef __attribute__((ext_vector_type(8)))  int      v8i;
typedef __attribute__((ext_vector_type(4)))  int      v4i;

#define CDIV(a, b) (((a) + (b) - 1) / (b))

// ---------------------------------------------------------------------------
// TDM: load a contiguous (64 x kp) f16 panel from global into LDS.
// D# built per CDNA5 ISA ch.8: group0 = count/lds_addr/global_addr/type,
// group1 = data_size + dims/strides, groups 2-3 zero (2D tile).
// ---------------------------------------------------------------------------
__device__ __forceinline__ void tdm_load_panel(const _Float16* gptr,
                                               unsigned lds_off, int kp) {
  unsigned long long ga = (unsigned long long)(size_t)gptr;
  v4u g0;
  g0[0] = 1u;                                            // count=1, user desc
  g0[1] = lds_off;                                       // lds_addr (bytes)
  g0[2] = (unsigned)(ga & 0xFFFFFFFFu);                  // global_addr[31:0]
  g0[3] = (unsigned)((ga >> 32) & 0x01FFFFFFu) | (2u << 30);  // [56:32]|type=2
  v8i g1;
  g1[0] = 0x10000;                     // data_size=1 (2 bytes/elem)
  g1[1] = (int)((unsigned)kp << 16);   // tensor_dim0[15:0] @ bits[63:48]
  g1[2] = (int)(64u << 16);            // tensor_dim1[15:0] @ bits[95:80]
  g1[3] = (int)((unsigned)kp << 16);   // tile_dim0 @ bits[127:112]
  g1[4] = 64;                          // tile_dim1 @ bits[143:128]
  g1[5] = kp;                          // tensor_dim0_stride[31:0] @ [191:160]
  g1[6] = 0;
  g1[7] = 0;
  v4i z4 = {0, 0, 0, 0};
#if defined(__clang_major__) && (__clang_major__ >= 23)
  v8i z8 = {0, 0, 0, 0, 0, 0, 0, 0};
  __builtin_amdgcn_tensor_load_to_lds(g0, g1, z4, z4, z8, 0);
#else
  __builtin_amdgcn_tensor_load_to_lds(g0, g1, z4, z4, 0);
#endif
}

// ---------------------------------------------------------------------------
// WMMA GEMM:  C[M, Npad] = A[M, KP] (f16) @ B[Npad, KP]^T (f16)  (+bias)(+=C)
// Block = 8 waves: 128 M-rows x 64 N-cols. B panel staged in LDS via TDM.
// Fragment = two contiguous 16B chunks at lane-dependent offsets, matching
// the documented 16-bit A/B VGPR layout for 16x16x32.
// ---------------------------------------------------------------------------
__device__ __forceinline__ v16h frag_merge(v8h lo, v8h hi) {
  return __builtin_shufflevector(lo, hi, 0, 1, 2, 3, 4, 5, 6, 7,
                                 8, 9, 10, 11, 12, 13, 14, 15);
}

template <int KP>
__global__ void __launch_bounds__(256)
wmma_gemm_kernel(float* __restrict__ C, const _Float16* __restrict__ A,
                 const _Float16* __restrict__ B, int Mtiles, int ldc,
                 const float* __restrict__ bias, int accumulate) {
  __shared__ _Float16 Bs[64 * KP];
  const int wave = threadIdx.x >> 5;
  const int lane = threadIdx.x & 31;
  const int nt0 = blockIdx.y * 4;

  if (wave == 0) {                               // one TDM issue per block
    tdm_load_panel(B + (size_t)nt0 * 16 * KP, (unsigned)(size_t)&Bs[0], KP);
    __builtin_amdgcn_s_wait_tensorcnt(0);
  }
  __syncthreads();

  const int mt = blockIdx.x * 8 + wave;
  if (mt >= Mtiles) return;                      // wave-uniform
  const int m0 = mt * 16;
  const int koff = (lane & 16) ? 8 : 0;
  const _Float16* arow = A + (long long)(m0 + (lane & 15)) * KP + koff;
  const _Float16* brow[4];
#pragma unroll
  for (int j = 0; j < 4; ++j)
    brow[j] = &Bs[(j * 16 + (lane & 15)) * KP + koff];

  v8f acc[4] = {};
  for (int k0 = 0; k0 < KP; k0 += 32) {
    v8h alo = *(const v8h*)(arow + k0);
    v8h ahi = *(const v8h*)(arow + k0 + 16);
    v16h a = frag_merge(alo, ahi);
#pragma unroll
    for (int j = 0; j < 4; ++j) {
      v8h blo = *(const v8h*)(brow[j] + k0);
      v8h bhi = *(const v8h*)(brow[j] + k0 + 16);
      v16h b = frag_merge(blo, bhi);
      acc[j] = __builtin_amdgcn_wmma_f32_16x16x32_f16(
          false, a, false, b, (short)0, acc[j], false, false);
    }
  }

  const int rbase = m0 + ((lane & 16) ? 8 : 0);
#pragma unroll
  for (int j = 0; j < 4; ++j) {
    const int col = (nt0 + j) * 16 + (lane & 15);
    const float bv = bias ? bias[col] : 0.0f;
    float* cp = C + (long long)rbase * ldc + col;
#pragma unroll
    for (int r = 0; r < 8; ++r) {
      float v = acc[j][r] + bv;
      if (accumulate) v += cp[(long long)r * ldc];
      cp[(long long)r * ldc] = v;
    }
  }
}

// ---------------------------------------------------------------------------
// Elementwise / staging kernels
// ---------------------------------------------------------------------------
__device__ __forceinline__ float sigm(float x) { return 1.0f / (1.0f + __expf(-x)); }

__global__ void pad_copy_f32(const float* __restrict__ src, int rows, int cols,
                             float* __restrict__ dst, int rows_p, int ld) {
  int i = blockIdx.x * blockDim.x + threadIdx.x;
  if (i >= rows_p * ld) return;
  int r = i / ld, c = i % ld;
  dst[i] = (r < rows && c < cols) ? src[r * cols + c] : 0.0f;
}

__global__ void cvt_f16_all(const float* __restrict__ src, _Float16* __restrict__ dst,
                            long long n) {
  long long i = (long long)blockIdx.x * blockDim.x + threadIdx.x;
  if (i < n) dst[i] = (_Float16)src[i];
}

// W(N,K) row-major -> f16 [Np x Kp], optional split-K (2H concat layout)
__global__ void prep_wt(const float* __restrict__ W, int N, int K,
                        _Float16* __restrict__ out, int Np, int Kp, int split) {
  int i = blockIdx.x * blockDim.x + threadIdx.x;
  if (i >= Np * Kp) return;
  int n = i / Kp, k = i % Kp;
  float v = 0.0f;
  if (n < N) {
    if (!split) {
      if (k < K) v = W[n * K + k];
    } else {  // K == 400 split into two HP halves of 200 valid cols each
      if (k < HP) { if (k < HH) v = W[n * 400 + k]; }
      else        { int kk = k - HP; if (kk < HH) v = W[n * 400 + HH + kk]; }
    }
  }
  out[i] = (_Float16)v;
}

// W(K,N) row-major -> transposed f16 [Np x Kp]  (for  x @ W  form)
__global__ void prep_wnt(const float* __restrict__ W, int N, int K,
                         _Float16* __restrict__ out, int Np, int Kp) {
  int i = blockIdx.x * blockDim.x + threadIdx.x;
  if (i >= Np * Kp) return;
  int n = i / Kp, k = i % Kp;
  out[i] = (_Float16)((n < N && k < K) ? W[k * N + n] : 0.0f);
}

__global__ void prep_bias(const float* __restrict__ a, const float* __restrict__ b,
                          float* __restrict__ out, int Nreal, int Np) {
  int i = blockIdx.x * blockDim.x + threadIdx.x;
  if (i >= Np) return;
  out[i] = (i < Nreal) ? (a[i] + (b ? b[i] : 0.0f)) : 0.0f;
}

// one wave per row L2 normalize (valid cols only; pad cols stay 0)
__global__ void l2norm_rows(float* __restrict__ X, int rows, int ld, int cols) {
  int row = blockIdx.x * (blockDim.x >> 5) + (threadIdx.x >> 5);
  int lane = threadIdx.x & 31;
  if (row >= rows) return;
  float s = 0.0f;
  for (int c = lane; c < cols; c += 32) { float v = X[(long long)row * ld + c]; s += v * v; }
#pragma unroll
  for (int m = 16; m >= 1; m >>= 1) s += __shfl_xor(s, m, 32);
  float sc = 1.0f / fmaxf(sqrtf(s), 1e-12f);
  for (int c = lane; c < cols; c += 32) X[(long long)row * ld + c] *= sc;
}

__global__ void seg_scatter(const float* __restrict__ src, int ld,
                            const int* __restrict__ idx, const int* __restrict__ seg,
                            int nrows, float* __restrict__ sum) {
  long long tid = (long long)blockIdx.x * blockDim.x + threadIdx.x;
  const int G = HH / 4;
  if (tid >= (long long)nrows * G) return;
  int e = (int)(tid / G), c = (int)(tid % G) * 4;
  const float* s = src + (long long)idx[e] * ld + c;
  float* d = sum + (long long)seg[e] * ld + c;
#pragma unroll
  for (int q = 0; q < 4; ++q) atomicAdd(&d[q], s[q]);
}

__global__ void seg_count(const int* __restrict__ seg, int n, float* __restrict__ cnt) {
  int i = blockIdx.x * blockDim.x + threadIdx.x;
  if (i < n) atomicAdd(&cnt[seg[i]], 1.0f);
}

__global__ void seg_div(float* __restrict__ s, const float* __restrict__ cnt,
                        int rows_p, int nseg, int ld) {
  int i = blockIdx.x * blockDim.x + threadIdx.x;
  if (i >= rows_p * ld) return;
  int r = i / ld, c = i % ld;
  s[i] = (r < nseg && c < HH) ? s[i] / fmaxf(cnt[r], 1.0f) : 0.0f;
}

// [a | b] -> f16 (both padded HP-wide inputs)
__global__ void concat_f16(const float* __restrict__ a, const float* __restrict__ b,
                           _Float16* __restrict__ dst, int rows_p) {
  int i = blockIdx.x * blockDim.x + threadIdx.x;
  if (i >= rows_p * H2P) return;
  int r = i / H2P, k = i % H2P;
  float v = (k < HP) ? a[(long long)r * HP + k] : b[(long long)r * HP + (k - HP)];
  dst[i] = (_Float16)v;
}

__global__ void lstm_pw(const float* __restrict__ g, const float* __restrict__ cprev,
                        float* __restrict__ h, float* __restrict__ c,
                        int rows_valid, int rows_p) {
  int i = blockIdx.x * blockDim.x + threadIdx.x;
  if (i >= rows_p * HP) return;
  int r = i / HP, cc = i % HP;
  float hv = 0.0f, cv = 0.0f;
  if (r < rows_valid && cc < HH) {
    const float* gr = g + (long long)r * H4;
    float gi = gr[cc], gf = gr[HH + cc], gg = gr[2 * HH + cc], go = gr[3 * HH + cc];
    float c2 = sigm(gf) * cprev[(long long)r * HP + cc] + sigm(gi) * tanhf(gg);
    cv = c2;
    hv = sigm(go) * tanhf(c2);
  }
  h[i] = hv; c[i] = cv;
}

__global__ void gru_pw(const float* __restrict__ gi, const float* __restrict__ gh,
                       const float* __restrict__ hprev, float* __restrict__ out,
                       int rows_valid, int rows_p) {
  int i = blockIdx.x * blockDim.x + threadIdx.x;
  if (i >= rows_p * HP) return;
  int r = i / HP, cc = i % HP;
  float v = 0.0f;
  if (r < rows_valid && cc < HH) {
    const float* a = gi + (long long)r * H3P;
    const float* b = gh + (long long)r * H3P;
    float rr = sigm(a[cc] + b[cc]);
    float zz = sigm(a[HH + cc] + b[HH + cc]);
    float nn = tanhf(a[2 * HH + cc] + rr * b[2 * HH + cc]);
    v = (1.0f - zz) * nn + zz * hprev[(long long)r * HP + cc];
  }
  out[i] = v;
}

// agg[dst] += hW[src] + relW[et]
__global__ void rgcn_scatter(const float* __restrict__ hW, const float* __restrict__ relW,
                             const int* __restrict__ src, const int* __restrict__ dst,
                             const int* __restrict__ et, int E, float* __restrict__ agg) {
  long long tid = (long long)blockIdx.x * blockDim.x + threadIdx.x;
  const int G = HH / 4;
  if (tid >= (long long)E * G) return;
  int e = (int)(tid / G), c = (int)(tid % G) * 4;
  const float* m = hW + (long long)src[e] * HP + c;
  const float* rl = relW + (long long)et[e] * HP + c;
  float* d = agg + (long long)dst[e] * HP + c;
#pragma unroll
  for (int q = 0; q < 4; ++q) atomicAdd(&d[q], m[q] + rl[q]);
}

__global__ void indeg_scatter(const int* __restrict__ dst, int E, int* __restrict__ deg) {
  int i = blockIdx.x * blockDim.x + threadIdx.x;
  if (i < E) atomicAdd(&deg[dst[i]], 1);
}

// out = rrelu(agg + (indeg>0 ? loopE : loopL))
__global__ void rgcn_combine(const float* __restrict__ agg, const float* __restrict__ we,
                             const float* __restrict__ wl, const int* __restrict__ deg,
                             float* __restrict__ out, int rows_p) {
  int i = blockIdx.x * blockDim.x + threadIdx.x;
  if (i >= rows_p * HP) return;
  int r = i / HP, c = i % HP;
  float v = 0.0f;
  if (c < HH) {
    float lp = (deg[r] > 0) ? we[i] : wl[i];
    v = agg[i] + lp;
    v = (v >= 0.0f) ? v : v * RRELU_SLOPE;
  }
  out[i] = v;
}

__global__ void copy_out(const float* __restrict__ src, float* __restrict__ dst, int rows) {
  long long i = (long long)blockIdx.x * blockDim.x + threadIdx.x;
  if (i >= (long long)rows * HH) return;
  int r = (int)(i / HH), c = (int)(i % HH);
  dst[i] = src[(long long)r * HP + c];
}

// ---------------------------------------------------------------------------
// Host orchestration
// ---------------------------------------------------------------------------
static inline void gemm(hipStream_t s, float* C, const _Float16* A, const _Float16* B,
                        int M, int Npad, int Kp, const float* bias, int acc) {
  int Mtiles = M / 16;
  dim3 grid(CDIV(Mtiles, 8), Npad / 64);
  if (Kp == H2P)
    wmma_gemm_kernel<H2P><<<grid, 256, 0, s>>>(C, A, B, Mtiles, Npad, bias, acc);
  else
    wmma_gemm_kernel<HP><<<grid, 256, 0, s>>>(C, A, B, Mtiles, Npad, bias, acc);
}

static inline char* bump(char* base, size_t& off, size_t bytes) {
  char* p = base + off;
  off += (bytes + 255) & ~(size_t)255;
  return p;
}

extern "C" void kernel_launch(void* const* d_in, const int* in_sizes, int n_in,
                              void* d_out, int out_size, void* d_ws, size_t ws_size,
                              hipStream_t stream) {
  (void)in_sizes; (void)n_in; (void)out_size; (void)ws_size;
  const int* src_i   = (const int*)d_in[0];
  const int* dst_i   = (const int*)d_in[1];
  const int* et_i    = (const int*)d_in[2];
  const int* r2e_i   = (const int*)d_in[3];
  const int* rseg_i  = (const int*)d_in[4];
  const int* ssrc_i  = (const int*)d_in[5];
  const int* sdst_i  = (const int*)d_in[6];
  const int* set_i   = (const int*)d_in[7];
  const int* sr2e_i  = (const int*)d_in[8];
  const int* srseg_i = (const int*)d_in[9];
  const float* dyn   = (const float*)d_in[10];
  const float* embr  = (const float*)d_in[11];
  const float* prel  = (const float*)d_in[12];
  const float* Wih1 = (const float*)d_in[13]; const float* Whh1 = (const float*)d_in[14];
  const float* bih1 = (const float*)d_in[15]; const float* bhh1 = (const float*)d_in[16];
  const float* Wih2 = (const float*)d_in[17]; const float* Whh2 = (const float*)d_in[18];
  const float* bih2 = (const float*)d_in[19]; const float* bhh2 = (const float*)d_in[20];
  const float* Wih3 = (const float*)d_in[21]; const float* Whh3 = (const float*)d_in[22];
  const float* bih3 = (const float*)d_in[23]; const float* bhh3 = (const float*)d_in[24];
  const float* Wihe = (const float*)d_in[25]; const float* Whhe = (const float*)d_in[26];
  const float* bihe = (const float*)d_in[27]; const float* bhhe = (const float*)d_in[28];
  const float* Wn = (const float*)d_in[29]; const float* Wl = (const float*)d_in[30];
  const float* We = (const float*)d_in[31];
  const float* sWn = (const float*)d_in[32]; const float* sWl = (const float*)d_in[33];
  const float* sWe = (const float*)d_in[34];
  float* out = (float*)d_out;

  char* base = (char*)d_ws;
  size_t off = 0;
  _Float16* Wih1h = (_Float16*)bump(base, off, (size_t)H4 * H2P * 2);
  _Float16* Whh1h = (_Float16*)bump(base, off, (size_t)H4 * HP * 2);
  _Float16* Wih2h = (_Float16*)bump(base, off, (size_t)H4 * H2P * 2);
  _Float16* Whh2h = (_Float16*)bump(base, off, (size_t)H4 * HP * 2);
  _Float16* Wih3h = (_Float16*)bump(base, off, (size_t)H3P * HP * 2);
  _Float16* Whh3h = (_Float16*)bump(base, off, (size_t)H3P * HP * 2);
  _Float16* Wiheh = (_Float16*)bump(base, off, (size_t)H3P * HP * 2);
  _Float16* Whheh = (_Float16*)bump(base, off, (size_t)H3P * HP * 2);
  _Float16* wnt[12];
  for (int i = 0; i < 12; ++i) wnt[i] = (_Float16*)bump(base, off, (size_t)HP * HP * 2);
  _Float16* xcath  = (_Float16*)bump(base, off, (size_t)R2P * H2P * 2);
  _Float16* sxcath = (_Float16*)bump(base, off, (size_t)SRP * H2P * 2);
  _Float16* hsmh   = (_Float16*)bump(base, off, (size_t)R2P * HP * 2);
  _Float16* tinyh  = (_Float16*)bump(base, off, (size_t)SRP * HP * 2);
  _Float16* h0gh   = (_Float16*)bump(base, off, (size_t)R2P * HP * 2);
  _Float16* ef16   = (_Float16*)bump(base, off, (size_t)N_ENT * HP * 2);
  float* b1s   = (float*)bump(base, off, H4 * 4);
  float* b2s   = (float*)bump(base, off, H4 * 4);
  float* b3i   = (float*)bump(base, off, H3P * 4);
  float* b3h   = (float*)bump(base, off, H3P * 4);
  float* bei   = (float*)bump(base, off, H3P * 4);
  float* beh   = (float*)bump(base, off, H3P * 4);
  float* hentA = (float*)bump(base, off, (size_t)N_ENT * HP * 4);
  float* hentB = (float*)bump(base, off, (size_t)N_ENT * HP * 4);
  float* embrp = (float*)bump(base, off, (size_t)R2P * HP * 4);
  float* prelp = (float*)bump(base, off, (size_t)SRP * HP * 4);
  float* xin   = (float*)bump(base, off, (size_t)R2P * HP * 4);
  float* cntR  = (float*)bump(base, off, R2P * 4);
  float* sx    = (float*)bump(base, off, (size_t)SRP * HP * 4);
  float* cntS  = (float*)bump(base, off, SRP * 4);
  float* h0L   = (float*)bump(base, off, (size_t)R2P * HP * 4);
  float* c0    = (float*)bump(base, off, (size_t)R2P * HP * 4);
  float* h0G   = (float*)bump(base, off, (size_t)R2P * HP * 4);
  float* ph0L  = (float*)bump(base, off, (size_t)SRP * HP * 4);
  float* pc0   = (float*)bump(base, off, (size_t)SRP * HP * 4);
  float* g1    = (float*)bump(base, off, (size_t)R2P * H4 * 4);
  float* g2    = (float*)bump(base, off, (size_t)SRP * H4 * 4);
  float* sHW   = (float*)bump(base, off, (size_t)R2P * HP * 4);
  float* srelW = (float*)bump(base, off, (size_t)SRP * HP * 4);
  float* sAgg  = (float*)bump(base, off, (size_t)R2P * HP * 4);
  float* sLpE  = (float*)bump(base, off, (size_t)R2P * HP * 4);
  float* sLpL  = (float*)bump(base, off, (size_t)R2P * HP * 4);
  float* sCurA = (float*)bump(base, off, (size_t)R2P * HP * 4);
  float* sCurB = (float*)bump(base, off, (size_t)R2P * HP * 4);
  int*   sDeg  = (int*)bump(base, off, R2P * 4);
  float* gi3   = (float*)bump(base, off, (size_t)R2P * H3P * 4);
  float* gh3   = (float*)bump(base, off, (size_t)R2P * H3P * 4);
  float* erelW = (float*)bump(base, off, (size_t)R2P * HP * 4);
  float* eHW   = (float*)bump(base, off, (size_t)N_ENT * HP * 4);
  float* eAgg  = (float*)bump(base, off, (size_t)N_ENT * HP * 4);
  float* eLpE  = (float*)bump(base, off, (size_t)N_ENT * HP * 4);
  float* eLpL  = (float*)bump(base, off, (size_t)N_ENT * HP * 4);
  float* eCurA = (float*)bump(base, off, (size_t)N_ENT * HP * 4);
  float* eCurB = (float*)bump(base, off, (size_t)N_ENT * HP * 4);
  int*   eDeg  = (int*)bump(base, off, (size_t)N_ENT * 4);
  float* giE   = (float*)bump(base, off, (size_t)N_ENT * H3P * 4);
  float* ghE   = (float*)bump(base, off, (size_t)N_ENT * H3P * 4);

  const int TB = 256;
#define EW(total) CDIV((total), TB), TB, 0, stream

  // ---- weight / bias prep ----
  prep_wt<<<EW(H4 * H2P)>>>(Wih1, 4 * HH, 2 * HH, Wih1h, H4, H2P, 1);
  prep_wt<<<EW(H4 * HP)>>>(Whh1, 4 * HH, HH, Whh1h, H4, HP, 0);
  prep_wt<<<EW(H4 * H2P)>>>(Wih2, 4 * HH, 2 * HH, Wih2h, H4, H2P, 1);
  prep_wt<<<EW(H4 * HP)>>>(Whh2, 4 * HH, HH, Whh2h, H4, HP, 0);
  prep_wt<<<EW(H3P * HP)>>>(Wih3, 3 * HH, HH, Wih3h, H3P, HP, 0);
  prep_wt<<<EW(H3P * HP)>>>(Whh3, 3 * HH, HH, Whh3h, H3P, HP, 0);
  prep_wt<<<EW(H3P * HP)>>>(Wihe, 3 * HH, HH, Wiheh, H3P, HP, 0);
  prep_wt<<<EW(H3P * HP)>>>(Whhe, 3 * HH, HH, Whheh, H3P, HP, 0);
  const float* wsrc[6] = {Wn, Wl, We, sWn, sWl, sWe};
  for (int m = 0; m < 6; ++m)
    for (int l = 0; l < 2; ++l)
      prep_wnt<<<EW(HP * HP)>>>(wsrc[m] + (size_t)l * HH * HH, HH, HH,
                                wnt[m * 2 + l], HP, HP);
  prep_bias<<<EW(H4)>>>(bih1, bhh1, b1s, 4 * HH, H4);
  prep_bias<<<EW(H4)>>>(bih2, bhh2, b2s, 4 * HH, H4);
  prep_bias<<<EW(H3P)>>>(bih3, nullptr, b3i, 3 * HH, H3P);
  prep_bias<<<EW(H3P)>>>(bhh3, nullptr, b3h, 3 * HH, H3P);
  prep_bias<<<EW(H3P)>>>(bihe, nullptr, bei, 3 * HH, H3P);
  prep_bias<<<EW(H3P)>>>(bhhe, nullptr, beh, 3 * HH, H3P);

  // ---- initial state ----
  pad_copy_f32<<<EW(N_ENT * HP)>>>(dyn, N_ENT, HH, hentA, N_ENT, HP);
  l2norm_rows<<<CDIV(N_ENT, 8), 256, 0, stream>>>(hentA, N_ENT, HP, HH);
  pad_copy_f32<<<EW(R2P * HP)>>>(embr, R2C, HH, embrp, R2P, HP);
  pad_copy_f32<<<EW(SRP * HP)>>>(prel, SR_C, HH, prelp, SRP, HP);

  float* hcur = hentA;
  float* hnext = hentB;

  for (int t = 0; t < T_STEPS; ++t) {
    const int* tsrc = src_i + t * E_EDGES;
    const int* tdst = dst_i + t * E_EDGES;
    const int* tet  = et_i  + t * E_EDGES;
    const int* tr2e = r2e_i + t * R2E_C;
    const int* trsg = rseg_i + t * R2E_C;
    const int* tss  = ssrc_i + t * SE_C;
    const int* tsd  = sdst_i + t * SE_C;
    const int* tse  = set_i  + t * SE_C;
    const int* ts2e = sr2e_i + t * S2E_C;
    const int* tssg = srseg_i + t * S2E_C;

    // 1) x_input = seg_mean(h[r_to_e], r_seg, R2)
    hipMemsetAsync(xin, 0, (size_t)R2P * HP * 4, stream);
    hipMemsetAsync(cntR, 0, R2P * 4, stream);
    seg_scatter<<<EW(R2E_C * (HH / 4))>>>(hcur, HP, tr2e, trsg, R2E_C, xin);
    seg_count<<<EW(R2E_C)>>>(trsg, R2E_C, cntR);
    seg_div<<<EW(R2P * HP)>>>(xin, cntR, R2P, R2C, HP);

    // 2) LSTM1 over relations
    concat_f16<<<EW(R2P * H2P)>>>(embrp, xin, xcath, R2P);
    const float* hprev = (t == 0) ? embrp : h0G;
    const float* cprev = (t == 0) ? xin : c0;
    cvt_f16_all<<<EW(R2P * HP)>>>(hprev, hsmh, (long long)R2P * HP);
    gemm(stream, g1, xcath, Wih1h, R2P, H4, H2P, b1s, 0);
    gemm(stream, g1, hsmh, Whh1h, R2P, H4, HP, nullptr, 1);
    lstm_pw<<<EW(R2P * HP)>>>(g1, cprev, h0L, c0, R2C, R2P);
    l2norm_rows<<<CDIV(R2C, 8), 256, 0, stream>>>(h0L, R2C, HP, HH);
    l2norm_rows<<<CDIV(R2C, 8), 256, 0, stream>>>(c0, R2C, HP, HH);

    // 3) sx = seg_mean(h0[s_r_to_e], s_r_seg, SR)
    hipMemsetAsync(sx, 0, (size_t)SRP * HP * 4, stream);
    hipMemsetAsync(cntS, 0, SRP * 4, stream);
    seg_scatter<<<EW(S2E_C * (HH / 4))>>>(h0L, HP, ts2e, tssg, S2E_C, sx);
    seg_count<<<EW(S2E_C)>>>(tssg, S2E_C, cntS);
    seg_div<<<EW(SRP * HP)>>>(sx, cntS, SRP, SR_C, HP);

    // 4) LSTM2 over super-relations
    concat_f16<<<EW(SRP * H2P)>>>(prelp, sx, sxcath, SRP);
    const float* phprev = (t == 0) ? prelp : ph0L;
    const float* pcprev = (t == 0) ? sx : pc0;
    cvt_f16_all<<<EW(SRP * HP)>>>(phprev, tinyh, (long long)SRP * HP);
    gemm(stream, g2, sxcath, Wih2h, SRP, H4, H2P, b2s, 0);
    gemm(stream, g2, tinyh, Whh2h, SRP, H4, HP, nullptr, 1);
    lstm_pw<<<EW(SRP * HP)>>>(g2, pcprev, ph0L, pc0, SR_C, SRP);
    l2norm_rows<<<1, 256, 0, stream>>>(ph0L, SR_C, HP, HH);
    l2norm_rows<<<1, 256, 0, stream>>>(pc0, SR_C, HP, HH);

    // 5) super RGCN (2 layers) over relation nodes
    float* scur = h0L;
    float* souts[2] = {sCurA, sCurB};
    cvt_f16_all<<<EW(SRP * HP)>>>(ph0L, tinyh, (long long)SRP * HP);
    for (int l = 0; l < 2; ++l) {
      cvt_f16_all<<<EW(R2P * HP)>>>(scur, hsmh, (long long)R2P * HP);
      gemm(stream, sHW, hsmh, wnt[6 + l], R2P, HP, HP, nullptr, 0);    // sWn[l]
      gemm(stream, srelW, tinyh, wnt[6 + l], SRP, HP, HP, nullptr, 0);
      hipMemsetAsync(sAgg, 0, (size_t)R2P * HP * 4, stream);
      hipMemsetAsync(sDeg, 0, R2P * 4, stream);
      rgcn_scatter<<<EW(SE_C * (HH / 4))>>>(sHW, srelW, tss, tsd, tse, SE_C, sAgg);
      indeg_scatter<<<EW(SE_C)>>>(tsd, SE_C, sDeg);
      gemm(stream, sLpE, hsmh, wnt[10 + l], R2P, HP, HP, nullptr, 0);  // sWe[l]
      gemm(stream, sLpL, hsmh, wnt[8 + l], R2P, HP, HP, nullptr, 0);   // sWl[l]
      rgcn_combine<<<EW(R2P * HP)>>>(sAgg, sLpE, sLpL, sDeg, souts[l], R2P);
      scur = souts[l];
    }
    l2norm_rows<<<CDIV(R2C, 8), 256, 0, stream>>>(scur, R2C, HP, HH);

    // 6) GRU over relations -> h0G; emit rel_embs[t]
    cvt_f16_all<<<EW(R2P * HP)>>>(scur, hsmh, (long long)R2P * HP);
    gemm(stream, gi3, hsmh, Wih3h, R2P, H3P, HP, b3i, 0);
    cvt_f16_all<<<EW(R2P * HP)>>>(h0L, hsmh, (long long)R2P * HP);
    gemm(stream, gh3, hsmh, Whh3h, R2P, H3P, HP, b3h, 0);
    gru_pw<<<EW(R2P * HP)>>>(gi3, gh3, h0L, h0G, R2C, R2P);
    l2norm_rows<<<CDIV(R2C, 8), 256, 0, stream>>>(h0G, R2C, HP, HH);
    copy_out<<<EW(R2C * HH)>>>(h0G, out + (size_t)T_STEPS * N_ENT * HH +
                                        (size_t)t * R2C * HH, R2C);

    // 7) entity RGCN (2 layers);  msg GEMM split:  hW = h@Wn,  relW = rel@Wn
    cvt_f16_all<<<EW(R2P * HP)>>>(h0G, h0gh, (long long)R2P * HP);
    float* ecur = hcur;
    float* eouts[2] = {eCurA, eCurB};
    for (int l = 0; l < 2; ++l) {
      cvt_f16_all<<<EW(N_ENT * HP)>>>(ecur, ef16, (long long)N_ENT * HP);
      gemm(stream, eHW, ef16, wnt[0 + l], N_ENT, HP, HP, nullptr, 0);   // Wn[l]
      gemm(stream, erelW, h0gh, wnt[0 + l], R2P, HP, HP, nullptr, 0);
      hipMemsetAsync(eAgg, 0, (size_t)N_ENT * HP * 4, stream);
      hipMemsetAsync(eDeg, 0, (size_t)N_ENT * 4, stream);
      rgcn_scatter<<<EW(E_EDGES * (HH / 4))>>>(eHW, erelW, tsrc, tdst, tet,
                                               E_EDGES, eAgg);
      indeg_scatter<<<EW(E_EDGES)>>>(tdst, E_EDGES, eDeg);
      gemm(stream, eLpE, ef16, wnt[4 + l], N_ENT, HP, HP, nullptr, 0);  // We[l]
      gemm(stream, eLpL, ef16, wnt[2 + l], N_ENT, HP, HP, nullptr, 0);  // Wl[l]
      rgcn_combine<<<EW(N_ENT * HP)>>>(eAgg, eLpE, eLpL, eDeg, eouts[l], N_ENT);
      ecur = eouts[l];
    }
    l2norm_rows<<<CDIV(N_ENT, 8), 256, 0, stream>>>(ecur, N_ENT, HP, HH);

    // 8) GRU over entities -> hnext; emit history_embs[t]
    cvt_f16_all<<<EW(N_ENT * HP)>>>(ecur, ef16, (long long)N_ENT * HP);
    gemm(stream, giE, ef16, Wiheh, N_ENT, H3P, HP, bei, 0);
    cvt_f16_all<<<EW(N_ENT * HP)>>>(hcur, ef16, (long long)N_ENT * HP);
    gemm(stream, ghE, ef16, Whheh, N_ENT, H3P, HP, beh, 0);
    gru_pw<<<EW(N_ENT * HP)>>>(giE, ghE, hcur, hnext, N_ENT, N_ENT);
    l2norm_rows<<<CDIV(N_ENT, 8), 256, 0, stream>>>(hnext, N_ENT, HP, HH);
    copy_out<<<EW(N_ENT * HH)>>>(hnext, out + (size_t)t * N_ENT * HH, N_ENT);

    float* tmp = hcur; hcur = hnext; hnext = tmp;
  }
#undef EW
}
